// DoorDetectionModel_38122129719499
// MI455X (gfx1250) — matrix operand order repaired and checked by reading, not compiled, not run on hardware
//
#include <hip/hip_runtime.h>
#include <hip/hip_bf16.h>

// ---------------------------------------------------------------------------
// MI455X (gfx1250) implementation of the EfficientNet-style door detector.
// 1x1 convs  -> bf16 WMMA GEMM (v_wmma_f32_16x16x32_bf16).
//               Block = 64M x 32N (4 waves). B tile (32K x 32N) staged into
//               LDS with GLOBAL_LOAD_ASYNC_TO_LDS_B32, double-buffered so the
//               async engine fills tile t+1 while WMMA consumes tile t.
//               Fused BN+SiLU epilogue.
// depthwise  -> direct VALU kernel (+BN+SiLU)
// SE         -> GAP + tiny MLP + broadcast scale
// edge map   -> gray + Sobel + threshold
// ---------------------------------------------------------------------------

typedef __attribute__((ext_vector_type(16))) __bf16   v16bf;
typedef __attribute__((ext_vector_type(8)))  float    v8f;
typedef __attribute__((ext_vector_type(8)))  unsigned v8u;

// Pack two floats into one dword of two bf16 (round-half-up + v_perm_b32).
__device__ __forceinline__ unsigned pack_bf16(float f0, float f1) {
    unsigned u0 = __builtin_bit_cast(unsigned, f0) + 0x8000u;
    unsigned u1 = __builtin_bit_cast(unsigned, f1) + 0x8000u;
    // result bytes: [0]=u0.b2 [1]=u0.b3 [2]=u1.b2 [3]=u1.b3
    return __builtin_amdgcn_perm(u1, u0, 0x07060302u);
}

__device__ __forceinline__ float silu(float y) {
    return y * (1.0f / (1.0f + __expf(-y)));
}

__device__ __forceinline__ v8f wmma_bf16(v8u a, v8u b, v8f c) {
    return __builtin_amdgcn_wmma_f32_16x16x32_bf16(
        false, __builtin_bit_cast(v16bf, a),
        false, __builtin_bit_cast(v16bf, b),
        (short)0, c, false, false);
}

// Async global -> LDS copy of one dword (per lane). Tracked by ASYNCcnt.
// VDST = per-lane LDS byte address (low 32 bits of generic LDS pointer).
__device__ __forceinline__ void async_ld_f32(const float* gp, float* lp) {
    unsigned lds = (unsigned)(size_t)lp;
    unsigned long long ga = (unsigned long long)(size_t)gp;
    asm volatile("global_load_async_to_lds_b32 %0, %1, off"
                 :: "v"(lds), "v"(ga) : "memory");
}

// ---------------------------------------------------------------------------
// 1x1 conv as GEMM:  out[b,m,hw] = act( g[m]*(sum_k W[m,k]*in[b,k,hw]) + bt[m] )
// B operand column n <-> (b = n/HW, hw = n%HW).
// Block: 128 thr = 4 waves; tile 64M x 32N. Each wave computes 32M x 16N
// (two A tiles share one B tile -> 2 WMMAs / K-step).
// B tile in LDS as bt[n][k] (fp32, stride 34) so K-pairs are ds_b64 reads.
// Double-buffered async staging; no early exits (barriers in loop).
// ---------------------------------------------------------------------------
#define BSTRIDE 34
__global__ void conv1x1_wmma_bnact(const float* __restrict__ in,
                                   const float* __restrict__ W,   // [M,K] row-major
                                   const float* __restrict__ g,
                                   const float* __restrict__ bt,
                                   float* __restrict__ out,
                                   int M, int K, int HW, int N, int act)
{
    __shared__ float sh[2][32 * BSTRIDE];       // double-buffered 32n x 32k tile

    const int tid  = threadIdx.x;
    const int lane = tid & 31;
    const int wave = tid >> 5;                  // 0..3
    const int half = lane >> 4;                 // K-half select (0/1)
    const int l16  = lane & 15;

    const int nblk = blockIdx.x * 32;           // block N origin
    const int mblk = blockIdx.y * 64;           // block M origin

    // ---- per-wave tile origins ----
    const int m0 = mblk + (wave & 1) * 32;      // 32 M rows per wave
    const int n0 = nblk + (wave >> 1) * 16;     // 16 N cols per wave

    // A rows (clamped: OOB rows compute garbage, never stored)
    const int am0 = m0 + l16;
    const int am1 = m0 + 16 + l16;
    const float* __restrict__ arow0 = W + (size_t)(am0 < M ? am0 : M - 1) * K;
    const float* __restrict__ arow1 = W + (size_t)(am1 < M ? am1 : M - 1) * K;

    // C column for this lane (clamped)
    const int bn = n0 + l16;
    const int cc = (bn < N) ? bn : N - 1;
    const int bb = cc / HW;
    const int bh = cc % HW;

    // ---- staging role: thread owns column nn, k rows kr+4i of the B tile ----
    const int nn = tid & 31;                    // 0..31
    const int kr = tid >> 5;                    // 0..3
    const int ng = nblk + nn;
    const int ngc = (ng < N) ? ng : N - 1;
    const float* __restrict__ colptr =
        in + ((size_t)(ngc / HW) * K) * HW + (ngc % HW);
    const int nl = (wave >> 1) * 16 + l16;      // column read back by this lane

    v8f acc0 = {}, acc1 = {};

    const int kfull = K & ~31;
    int p = 0;
    if (kfull > 0) {          // prologue: stage tile 0 into buffer 0
#pragma unroll
        for (int i = 0; i < 8; ++i) {
            const int kk = kr + i * 4;
            async_ld_f32(colptr + (size_t)kk * HW, &sh[0][nn * BSTRIDE + kk]);
        }
    }
    for (int k0 = 0; k0 < kfull; k0 += 32) {
        __builtin_prefetch(colptr + (size_t)(k0 + 64) * HW, 0, 0);
        if (k0 + 32 < kfull) {
            // stage next tile into the spare buffer while this one computes
#pragma unroll
            for (int i = 0; i < 8; ++i) {
                const int kk = kr + i * 4;
                async_ld_f32(colptr + (size_t)(k0 + 32 + kk) * HW,
                             &sh[p ^ 1][nn * BSTRIDE + kk]);
            }
            // in-order completion: <=8 outstanding => previous tile landed
            asm volatile("s_wait_asynccnt 0x8" ::: "memory");
        } else {
            asm volatile("s_wait_asynccnt 0x0" ::: "memory");
        }
        __syncthreads();       // all waves: buffer p fully written
        // ---- fragments + WMMA from buffer p
        v8u a0, a1, b;
#pragma unroll
        for (int v = 0; v < 8; ++v) {
            const int kb = k0 + ((v < 4) ? (v * 2) : (16 + (v - 4) * 2)) + 8 * half;
            float2 fa0 = *(const float2*)(arow0 + kb);
            float2 fa1 = *(const float2*)(arow1 + kb);
            a0[v] = pack_bf16(fa0.x, fa0.y);
            a1[v] = pack_bf16(fa1.x, fa1.y);
            const float2 fb = *(const float2*)&sh[p][nl * BSTRIDE + ((v < 4) ? (v * 2) : (16 + (v - 4) * 2)) + 8 * half];
            b[v] = pack_bf16(fb.x, fb.y);
        }
        acc0 = wmma_bf16(a0, b, acc0);
        acc1 = wmma_bf16(a1, b, acc1);
        __syncthreads();       // all waves done reading p -> may overwrite next iter
        p ^= 1;
    }

    if (kfull < K) {   // branchless tail: synchronous zero-filled staging
#pragma unroll
        for (int i = 0; i < 8; ++i) {
            const int kk = kr + i * 4;
            const int kg = kfull + kk;
            const int kgc = (kg < K) ? kg : K - 1;
            float vv = colptr[(size_t)kgc * HW];
            sh[p][nn * BSTRIDE + kk] = (kg < K) ? vv : 0.0f;
        }
        __syncthreads();
        v8u a0, a1, b;
#pragma unroll
        for (int v = 0; v < 8; ++v) {
            const int kb  = kfull + ((v < 4) ? (v * 2) : (16 + (v - 4) * 2)) + 8 * half;
            const int kb0 = (kb     < K) ? kb     : 0;
            const int kb1 = (kb + 1 < K) ? kb + 1 : 0;
            float f00 = arow0[kb0], f01 = arow0[kb1];
            float f10 = arow1[kb0], f11 = arow1[kb1];
            a0[v] = pack_bf16((kb < K) ? f00 : 0.0f, (kb + 1 < K) ? f01 : 0.0f);
            a1[v] = pack_bf16((kb < K) ? f10 : 0.0f, (kb + 1 < K) ? f11 : 0.0f);
            const float2 fb = *(const float2*)&sh[p][nl * BSTRIDE + ((v < 4) ? (v * 2) : (16 + (v - 4) * 2)) + 8 * half];
            b[v] = pack_bf16(fb.x, fb.y);   // zeros already staged for k >= K
        }
        acc0 = wmma_bf16(a0, b, acc0);
        acc1 = wmma_bf16(a1, b, acc1);
    }

    // ---- fused BN (+SiLU) epilogue; C layout: row = tile_m0 + r + 8*half
    if (bn < N) {
        float* __restrict__ ob = out + (size_t)bb * M * HW + bh;
#pragma unroll
        for (int r = 0; r < 8; ++r) {
            const int m = m0 + r + 8 * half;
            if (m < M) {
                float y = acc0[r] * g[m] + bt[m];
                if (act) y = silu(y);
                ob[(size_t)m * HW] = y;
            }
            const int m2 = m0 + 16 + r + 8 * half;
            if (m2 < M) {
                float y = acc1[r] * g[m2] + bt[m2];
                if (act) y = silu(y);
                ob[(size_t)m2 * HW] = y;
            }
        }
    }
}

// ---------------------------------------------------------------------------
// Depthwise kxk conv + BN + SiLU (SAME padding), NCHW.
// ---------------------------------------------------------------------------
__global__ void dwconv_bnact(const float* __restrict__ in,
                             const float* __restrict__ w,    // [C,1,k,k]
                             const float* __restrict__ g,
                             const float* __restrict__ bt,
                             float* __restrict__ out,
                             int C, int Hin, int Hout, int k, int stride,
                             int pad, int Bn)
{
    long long idx = (long long)blockIdx.x * blockDim.x + threadIdx.x;
    long long total = (long long)Bn * C * Hout * Hout;
    if (idx >= total) return;
    int wo = (int)(idx % Hout); long long t = idx / Hout;
    int ho = (int)(t % Hout);   t /= Hout;
    int c  = (int)(t % C);
    int b  = (int)(t / C);

    const float* ip = in + ((size_t)b * C + c) * Hin * Hin;
    const float* wp = w + (size_t)c * k * k;
    float acc = 0.0f;
    for (int ky = 0; ky < k; ++ky) {
        int hi = ho * stride - pad + ky;
        if (hi < 0 || hi >= Hin) continue;
        for (int kx = 0; kx < k; ++kx) {
            int wi = wo * stride - pad + kx;
            if (wi < 0 || wi >= Hin) continue;
            acc += ip[hi * Hin + wi] * wp[ky * k + kx];
        }
    }
    float y = acc * g[c] + bt[c];
    out[idx] = silu(y);
}

// ---------------------------------------------------------------------------
// Generic small direct conv (stem: 4->32, 3x3, stride 2) + BN + SiLU.
// ---------------------------------------------------------------------------
__global__ void conv_direct_bnact(const float* __restrict__ in,
                                  const float* __restrict__ w,  // [Cout,Cin,k,k]
                                  const float* __restrict__ g,
                                  const float* __restrict__ bt,
                                  float* __restrict__ out,
                                  int Cin, int Cout, int Hin, int Hout,
                                  int k, int stride, int pad, int Bn)
{
    long long idx = (long long)blockIdx.x * blockDim.x + threadIdx.x;
    long long total = (long long)Bn * Cout * Hout * Hout;
    if (idx >= total) return;
    int wo = (int)(idx % Hout); long long t = idx / Hout;
    int ho = (int)(t % Hout);   t /= Hout;
    int co = (int)(t % Cout);
    int b  = (int)(t / Cout);

    const float* ib = in + (size_t)b * Cin * Hin * Hin;
    const float* wb = w + (size_t)co * Cin * k * k;
    float acc = 0.0f;
    for (int ci = 0; ci < Cin; ++ci) {
        const float* ip = ib + (size_t)ci * Hin * Hin;
        const float* wp = wb + ci * k * k;
        for (int ky = 0; ky < k; ++ky) {
            int hi = ho * stride - pad + ky;
            if (hi < 0 || hi >= Hin) continue;
            for (int kx = 0; kx < k; ++kx) {
                int wi = wo * stride - pad + kx;
                if (wi < 0 || wi >= Hin) continue;
                acc += ip[hi * Hin + wi] * wp[ky * k + kx];
            }
        }
    }
    float y = acc * g[co] + bt[co];
    out[idx] = silu(y);
}

// ---------------------------------------------------------------------------
// Edge-map path: gray + copy x into concat[0..2]; then Sobel -> concat[3].
// ---------------------------------------------------------------------------
__global__ void gray_concat(const float* __restrict__ x, float* __restrict__ cc,
                            float* __restrict__ gray, int Bn, int H, int W)
{
    long long idx = (long long)blockIdx.x * blockDim.x + threadIdx.x;
    long long total = (long long)Bn * H * W;
    if (idx >= total) return;
    int wcol = (int)(idx % W); long long t = idx / W;
    int h = (int)(t % H);
    int b = (int)(t / H);
    size_t hw = (size_t)h * W + wcol;
    size_t xb = (size_t)b * 3 * H * W;
    float r = x[xb + hw], gch = x[xb + (size_t)H * W + hw], bl = x[xb + 2ull * H * W + hw];
    size_t cb = (size_t)b * 4 * H * W;
    cc[cb + hw] = r;
    cc[cb + (size_t)H * W + hw] = gch;
    cc[cb + 2ull * H * W + hw] = bl;
    gray[(size_t)b * H * W + hw] = 0.299f * r + 0.587f * gch + 0.114f * bl;
}

__global__ void sobel_edge(const float* __restrict__ gray, float* __restrict__ cc,
                           int Bn, int H, int W)
{
    long long idx = (long long)blockIdx.x * blockDim.x + threadIdx.x;
    long long total = (long long)Bn * H * W;
    if (idx >= total) return;
    int wcol = (int)(idx % W); long long t = idx / W;
    int h = (int)(t % H);
    int b = (int)(t / H);
    const float* gp = gray + (size_t)b * H * W;
    const float KX[3][3] = {{-1.f, 0.f, 1.f}, {-2.f, 0.f, 2.f}, {-1.f, 0.f, 1.f}};
    float gx = 0.f, gy = 0.f;
    for (int ky = 0; ky < 3; ++ky) {
        int hi = h - 1 + ky;
        if (hi < 0 || hi >= H) continue;
        for (int kx = 0; kx < 3; ++kx) {
            int wi = wcol - 1 + kx;
            if (wi < 0 || wi >= W) continue;
            float v = gp[(size_t)hi * W + wi];
            gx += KX[ky][kx] * v;
            gy += KX[kx][ky] * v;   // kx.T
        }
    }
    float mag = sqrtf(gx * gx + gy * gy + 1e-12f);
    cc[(size_t)b * 4 * H * W + 3ull * H * W + (size_t)h * W + wcol] =
        (mag * 255.0f > 200.0f) ? 1.0f : 0.0f;
}

// ---------------------------------------------------------------------------
// Global average pool over HW: one block per (b,c).
// ---------------------------------------------------------------------------
__global__ void gap_kernel(const float* __restrict__ in, float* __restrict__ out, int HW)
{
    __shared__ float red[128];
    const float* p = in + (size_t)blockIdx.x * HW;
    float s = 0.f;
    for (int i = threadIdx.x; i < HW; i += blockDim.x) s += p[i];
    red[threadIdx.x] = s;
    __syncthreads();
    for (int o = 64; o > 0; o >>= 1) {
        if ((int)threadIdx.x < o) red[threadIdx.x] += red[threadIdx.x + o];
        __syncthreads();
    }
    if (threadIdx.x == 0) out[blockIdx.x] = red[0] / (float)HW;
}

// ---------------------------------------------------------------------------
// SE MLP: gate[b,e] = sigmoid(sew @ silu(srw @ pool + srb) + seb)
// ---------------------------------------------------------------------------
__global__ void se_mlp(const float* __restrict__ pool, const float* __restrict__ srw,
                       const float* __restrict__ srb, const float* __restrict__ sew,
                       const float* __restrict__ seb, float* __restrict__ gate,
                       int E, int R)
{
    __shared__ float shp[720];
    __shared__ float sh1[192];
    int b = blockIdx.x;
    const float* p = pool + (size_t)b * E;
    for (int i = threadIdx.x; i < E; i += blockDim.x) shp[i] = p[i];
    __syncthreads();
    for (int r = threadIdx.x; r < R; r += blockDim.x) {
        float s = srb[r];
        const float* wr = srw + (size_t)r * E;
        for (int e = 0; e < E; ++e) s += wr[e] * shp[e];
        sh1[r] = silu(s);
    }
    __syncthreads();
    for (int e = threadIdx.x; e < E; e += blockDim.x) {
        float s = seb[e];
        const float* we = sew + (size_t)e * R;
        for (int r = 0; r < R; ++r) s += we[r] * sh1[r];
        gate[(size_t)b * E + e] = 1.0f / (1.0f + __expf(-s));
    }
}

__global__ void se_scale(float* __restrict__ x, const float* __restrict__ gate,
                         int C, int HW, long long total)
{
    long long i = (long long)blockIdx.x * blockDim.x + threadIdx.x;
    if (i >= total) return;
    int c = (int)((i / HW) % C);
    int b = (int)(i / ((long long)C * HW));
    x[i] *= gate[(size_t)b * C + c];
}

// ---------------------------------------------------------------------------
// Final FC: out[b,c] = pooled[b,:] . fcw[c,:] + fcb[c]
// ---------------------------------------------------------------------------
__global__ void fc_kernel(const float* __restrict__ pooled, const float* __restrict__ fw,
                          const float* __restrict__ fb, float* __restrict__ out,
                          int Bn, int C, int NC)
{
    int idx = blockIdx.x * blockDim.x + threadIdx.x;
    if (idx >= Bn * NC) return;
    int b = idx / NC, c = idx % NC;
    const float* p = pooled + (size_t)b * C;
    const float* wv = fw + (size_t)c * C;
    float s = fb[c];
    for (int i = 0; i < C; ++i) s += p[i] * wv[i];
    out[idx] = s;
}

// ---------------------------------------------------------------------------
// Host orchestration
// ---------------------------------------------------------------------------
static inline int cdiv(long long a, long long b) { return (int)((a + b - 1) / b); }

extern "C" void kernel_launch(void* const* d_in, const int* in_sizes, int n_in,
                              void* d_out, int out_size, void* d_ws, size_t ws_size,
                              hipStream_t stream)
{
    (void)in_sizes; (void)n_in; (void)out_size; (void)ws_size;
#define P(i) ((const float*)d_in[(i)])
    // jax-pytree (sorted-key) flatten order: params{blocks,head,stem}, then x.
    // block0 (no expand): db,dg,dw,pb,pg,pw,seb,sew,srb,srw        -> base 0
    // blocks1-4:          db,dg,dw,eb,eg,ew,pb,pg,pw,seb,sew,srb,srw
    const int B0 = 0, B1 = 10, B2 = 23, B3 = 36, B4 = 49;
    const int HEAD = 62;  // b, fc_b, fc_w, g, w
    const int STEM = 67;  // b, g, w
    const float* x = P(70);

    const int Bn = 32;

    // ---- workspace carving (floats) ----
    float* wsf = (float*)d_ws;
    size_t off = 0;
    float* concat = wsf + off; off += (size_t)Bn * 4 * 260 * 260;
    float* gray   = wsf + off; off += (size_t)Bn * 260 * 260;
    const size_t BIG = (size_t)Bn * 96 * 130 * 130;   // largest activation
    float* bufA   = wsf + off; off += BIG;
    float* bufB   = wsf + off; off += BIG;
    float* pool   = wsf + off; off += (size_t)Bn * 1408;
    float* gate   = wsf + off; off += (size_t)Bn * 720;

    // ---- edge map + concat ----
    {
        long long tot = (long long)Bn * 260 * 260;
        gray_concat<<<cdiv(tot, 256), 256, 0, stream>>>(x, concat, gray, Bn, 260, 260);
        sobel_edge <<<cdiv(tot, 256), 256, 0, stream>>>(gray, concat, Bn, 260, 260);
    }

    // ---- stem: 4->32, 3x3 s2, SAME (pad_low=0), 260 -> 130 ----
    {
        long long tot = (long long)Bn * 32 * 130 * 130;
        conv_direct_bnact<<<cdiv(tot, 256), 256, 0, stream>>>(
            concat, P(STEM + 2), P(STEM + 1), P(STEM + 0), bufA,
            4, 32, 260, 130, 3, 2, 0, Bn);
    }

    float* cur = bufA;
    float* nxt = bufB;

    struct Blk { int base; int has_e; int cin, E, R, cout, k, stride, hin, hout, pad; };
    const Blk blks[5] = {
        { B0, 0,  32,  32,   8,  16, 3, 1, 130, 130, 1 },
        { B1, 1,  16,  96,  24,  24, 3, 2, 130,  65, 0 },
        { B2, 1,  24, 144,  36,  48, 5, 2,  65,  33, 2 },
        { B3, 1,  48, 288,  72, 120, 3, 2,  33,  17, 1 },
        { B4, 1, 120, 720, 180, 352, 5, 2,  17,   9, 2 },
    };

    for (int i = 0; i < 5; ++i) {
        const Blk& bk = blks[i];
        const int hwin = bk.hin * bk.hin, hwout = bk.hout * bk.hout;
        const float *db, *dg, *dw, *pb, *pg, *pw, *seb, *sew, *srb, *srw;
        db = P(bk.base + 0); dg = P(bk.base + 1); dw = P(bk.base + 2);
        if (bk.has_e) {
            const float* eb = P(bk.base + 3);
            const float* eg = P(bk.base + 4);
            const float* ew = P(bk.base + 5);
            pb = P(bk.base + 6);  pg = P(bk.base + 7);  pw = P(bk.base + 8);
            seb = P(bk.base + 9); sew = P(bk.base + 10);
            srb = P(bk.base + 11); srw = P(bk.base + 12);
            // 1x1 expand: cin -> E @ hin, SiLU
            int N = Bn * hwin;
            dim3 grid(cdiv(N, 32), cdiv(bk.E, 64));
            conv1x1_wmma_bnact<<<grid, 128, 0, stream>>>(
                cur, ew, eg, eb, nxt, bk.E, bk.cin, hwin, N, 1);
            float* t = cur; cur = nxt; nxt = t;
        } else {
            pb = P(bk.base + 3);  pg = P(bk.base + 4);  pw = P(bk.base + 5);
            seb = P(bk.base + 6); sew = P(bk.base + 7);
            srb = P(bk.base + 8); srw = P(bk.base + 9);
        }
        // depthwise kxk stride s, SiLU
        {
            long long tot = (long long)Bn * bk.E * hwout;
            dwconv_bnact<<<cdiv(tot, 256), 256, 0, stream>>>(
                cur, dw, dg, db, nxt, bk.E, bk.hin, bk.hout, bk.k, bk.stride, bk.pad, Bn);
            float* t = cur; cur = nxt; nxt = t;
        }
        // SE
        gap_kernel<<<Bn * bk.E, 128, 0, stream>>>(cur, pool, hwout);
        se_mlp<<<Bn, 256, 0, stream>>>(pool, srw, srb, sew, seb, gate, bk.E, bk.R);
        {
            long long tot = (long long)Bn * bk.E * hwout;
            se_scale<<<cdiv(tot, 256), 256, 0, stream>>>(cur, gate, bk.E, hwout, tot);
        }
        // 1x1 project: E -> cout @ hout, no activation
        {
            int N = Bn * hwout;
            dim3 grid(cdiv(N, 32), cdiv(bk.cout, 64));
            conv1x1_wmma_bnact<<<grid, 128, 0, stream>>>(
                cur, pw, pg, pb, nxt, bk.cout, bk.E, hwout, N, 0);
            float* t = cur; cur = nxt; nxt = t;
        }
    }

    // ---- head 1x1: 352 -> 1408 @ 9x9, SiLU ----
    {
        int N = Bn * 81;
        dim3 grid(cdiv(N, 32), cdiv(1408, 64));
        conv1x1_wmma_bnact<<<grid, 128, 0, stream>>>(
            cur, P(HEAD + 4), P(HEAD + 3), P(HEAD + 0), nxt, 1408, 352, 81, N, 1);
        float* t = cur; cur = nxt; nxt = t;
    }
    // ---- global average pool + FC ----
    gap_kernel<<<Bn * 1408, 128, 0, stream>>>(cur, pool, 81);
    fc_kernel<<<1, 64, 0, stream>>>(pool, P(HEAD + 2), P(HEAD + 1), (float*)d_out,
                                    Bn, 1408, 2);
#undef P
}
